// GAT_75788992906285
// MI455X (gfx1250) — compile-verified, matrix operand order checked by reading, not run on hardware
//
#include <hip/hip_runtime.h>
#include <hip/hip_bf16.h>
#include <stdint.h>

// ---------------------------------------------------------------------------
// CDNA5 (gfx1250) GAT implementation, round 2.
//  - GEMMs: wave32 WMMA (v_wmma_f32_16x16x32_bf16), K fully unrolled.
//    * A tile staged through LDS (shared by 4 waves/block), bf16-converted once.
//    * B pre-transposed to N x K bf16 so each fragment is two aligned b128 loads.
//  - Edge phase: streaming grid-stride kernels, f32 global atomics (51MB
//    accumulator lives in MI455X's 192MB L2).
// ---------------------------------------------------------------------------

typedef __attribute__((ext_vector_type(16))) __bf16    v16bf;
typedef __attribute__((ext_vector_type(8)))  float     v8f;
typedef __attribute__((ext_vector_type(4)))  float     v4f;
typedef __attribute__((ext_vector_type(4)))  uint32_t  v4u;
typedef __attribute__((ext_vector_type(8)))  uint32_t  v8u;
typedef __attribute__((ext_vector_type(2)))  uint32_t  v2u;

static __device__ __forceinline__ uint16_t f2bf_bits(float f) {
  union { float f; uint32_t u; } v; v.f = f;
  uint32_t r = v.u + 0x7FFFu + ((v.u >> 16) & 1u);   // round-to-nearest-even
  return (uint16_t)(r >> 16);
}
static __device__ __forceinline__ __bf16 f2bf(float f) {
  return __builtin_bit_cast(__bf16, f2bf_bits(f));
}

static __device__ __forceinline__ void atomicMaxF(float* addr, float v) {
  if (v >= 0.0f) atomicMax((int*)addr, __float_as_int(v));
  else           atomicMin((unsigned int*)addr, __float_as_uint(v));
}

// ---------------------------------------------------------------------------
__global__ void fill_f32(float* __restrict__ p, float v, long long n) {
  long long i = (long long)blockIdx.x * blockDim.x + threadIdx.x;
  long long stride = (long long)gridDim.x * blockDim.x;
  for (; i < n; i += stride) p[i] = v;
}

// Wt[n*K + k] = bf16(W[k*N + n])   (transpose + convert)
__global__ void cvt_transpose_bf16(const float* __restrict__ W, __bf16* __restrict__ Wt,
                                   int K, int N) {
  int i = blockIdx.x * blockDim.x + threadIdx.x;     // over K*N
  if (i < K * N) {
    int k = i / N, n = i % N;
    Wt[(long long)n * K + k] = f2bf(W[i]);
  }
}

// ---------------------------------------------------------------------------
// C[M,N] = A[M,K]_f32 x Bt[N,K]_bf16^T, f32 accumulate via WMMA.
// Block: (32, WAVES). Each wave owns one 16x16 tile; all waves share A rows.
// Requires M%16==0, N==16*gridDim.x*blockDim.y, K%32==0.
template <int KC>
__global__ __launch_bounds__(128)
void gemm_wmma_bf16(const float* __restrict__ A, const __bf16* __restrict__ Bt,
                    float* __restrict__ C, int M, int N, int Krt) {
  const int K = KC ? KC : Krt;
  __shared__ alignas(16) uint32_t AsU[16 * 20];      // 16 rows x 32 bf16, stride 20 dw

  const int lane = threadIdx.x;                      // 0..31 (wave32)
  const int tn   = blockIdx.x * blockDim.y + threadIdx.y;
  const int tm   = blockIdx.y;
  const int half = lane >> 4;
  const int loc  = lane & 15;
  const int tid      = threadIdx.y * 32 + lane;
  const int nthreads = 32 * blockDim.y;
  const long long colK = (long long)(tn * 16 + loc) * K;   // row of Bt for this lane

  v8f acc = {};
#pragma unroll
  for (int k0 = 0; k0 < K; k0 += 32) {
    __syncthreads();                                 // previous fragments consumed
    // ---- cooperative A tile (16x32 f32 -> bf16) into LDS ----
    for (int q = tid; q < 128; q += nthreads) {      // 128 float4s
      const int row = q >> 3, c = q & 7;
      v4f f = *(const v4f*)(A + (long long)(tm * 16 + row) * K + k0 + c * 4);
      v2u pk;
      pk[0] = (uint32_t)f2bf_bits(f[0]) | ((uint32_t)f2bf_bits(f[1]) << 16);
      pk[1] = (uint32_t)f2bf_bits(f[2]) | ((uint32_t)f2bf_bits(f[3]) << 16);
      *(v2u*)&AsU[row * 20 + c * 2] = pk;            // ds_store_b64
    }
    __syncthreads();

    // ---- A fragment: two contiguous 8-elem K runs (ISA 7.12.2 16-bit layout) ----
    v4u ar0 = *(const v4u*)&AsU[loc * 20 + 4 * half];        // K = k0+8h .. +7
    v4u ar1 = *(const v4u*)&AsU[loc * 20 + 8 + 4 * half];    // K = k0+16+8h .. +7
    // ---- B fragment: symmetric layout, contiguous in transposed Bt ----
    v4u br0 = *(const v4u*)(Bt + colK + k0 + 8 * half);
    v4u br1 = *(const v4u*)(Bt + colK + k0 + 16 + 8 * half);

    v8u ap, bp;
#pragma unroll
    for (int j = 0; j < 4; ++j) {
      ap[j] = ar0[j]; ap[4 + j] = ar1[j];
      bp[j] = br0[j]; bp[4 + j] = br1[j];
    }
    acc = __builtin_amdgcn_wmma_f32_16x16x32_bf16(
        false, __builtin_bit_cast(v16bf, ap),
        false, __builtin_bit_cast(v16bf, bp),
        (short)0, acc, false, false);
  }
#pragma unroll
  for (int r = 0; r < 8; ++r)    // D: VGPR r -> M = r + 8*half, N = lane&15
    C[(long long)(tm * 16 + r + 8 * half) * N + tn * 16 + loc] = acc[r];
}

// ---------------------------------------------------------------------------
// a_src[n,h] = sum_c h[n,h,c]*att_src[h,c]  (and same for a_dst)
__global__ void attn_coef(const float* __restrict__ h, const float* __restrict__ att_s,
                          const float* __restrict__ att_d, float* __restrict__ as_out,
                          float* __restrict__ ad_out, long long Nn, int H, int C) {
  long long i = (long long)blockIdx.x * blockDim.x + threadIdx.x;
  long long stride = (long long)gridDim.x * blockDim.x;
  long long total = Nn * H;
  for (; i < total; i += stride) {
    long long n = i / H; int hh = (int)(i % H);
    const float* hp = h + n * H * C + (long long)hh * C;
    const float* sp = att_s + hh * C;
    const float* dp = att_d + hh * C;
    float ss = 0.0f, dd = 0.0f;
    for (int c = 0; c < C; ++c) { float v = hp[c]; ss += v * sp[c]; dd += v * dp[c]; }
    as_out[i] = ss; ad_out[i] = dd;
  }
}

// ---------------------------------------------------------------------------
// Pass 1: e = leaky_relu(a_src[src]+a_dst[dst]); store, segment-max into m[dst].
__global__ void edge_logits(const int* __restrict__ ei, long long E, long long Etot,
                            const float* __restrict__ as, const float* __restrict__ ad,
                            float* __restrict__ ews, float* __restrict__ m, int H) {
  long long i = (long long)blockIdx.x * blockDim.x + threadIdx.x;
  long long stride = (long long)gridDim.x * blockDim.x;
  long long total = Etot * H;
  for (; i < total; i += stride) {
    long long e = i / H; int hh = (int)(i % H);
    int s = (e < E) ? ei[e]     : (int)(e - E);
    int d = (e < E) ? ei[E + e] : (int)(e - E);
    float v = as[(long long)s * H + hh] + ad[(long long)d * H + hh];
    v = (v > 0.0f) ? v : 0.2f * v;                   // LEAKY_SLOPE = 0.2
    ews[i] = v;
    atomicMaxF(&m[(long long)d * H + hh], v);
  }
}

// Pass 2: ex = exp(e - m[dst]); store back, segment-sum into s[dst].
__global__ void edge_expsum(const int* __restrict__ ei, long long E, long long Etot,
                            const float* __restrict__ m, float* __restrict__ ews,
                            float* __restrict__ ssum, int H) {
  long long i = (long long)blockIdx.x * blockDim.x + threadIdx.x;
  long long stride = (long long)gridDim.x * blockDim.x;
  long long total = Etot * H;
  for (; i < total; i += stride) {
    long long e = i / H; int hh = (int)(i % H);
    int d = (e < E) ? ei[E + e] : (int)(e - E);
    float ex = __expf(ews[i] - m[(long long)d * H + hh]);
    ews[i] = ex;
    atomicAdd(&ssum[(long long)d * H + hh], ex);
  }
}

// Pass 3: agg[dst, j] += h[src, j] * alpha(e, h)   (j fastest -> coalesced)
__global__ void edge_aggregate(const int* __restrict__ ei, long long E, long long Etot,
                               const float* __restrict__ hfeat, const float* __restrict__ ews,
                               const float* __restrict__ ssum, float* __restrict__ agg,
                               int H, int C) {
  const int HC = H * C;
  long long i = (long long)blockIdx.x * blockDim.x + threadIdx.x;
  long long stride = (long long)gridDim.x * blockDim.x;
  long long total = Etot * HC;
  for (; i < total; i += stride) {
    long long e = i / HC; int j = (int)(i % HC); int hh = j / C;
    int s = (e < E) ? ei[e]     : (int)(e - E);
    int d = (e < E) ? ei[E + e] : (int)(e - E);
    float alpha = ews[e * H + hh] / (ssum[(long long)d * H + hh] + 1e-16f);
    atomicAdd(&agg[(long long)d * HC + j], hfeat[(long long)s * HC + j] * alpha);
  }
}

// ---------------------------------------------------------------------------
__global__ void bias_elu_inplace(float* __restrict__ x, const float* __restrict__ b,
                                 long long n, int F) {
  long long i = (long long)blockIdx.x * blockDim.x + threadIdx.x;
  long long stride = (long long)gridDim.x * blockDim.x;
  for (; i < n; i += stride) {
    float v = x[i] + b[i % F];
    x[i] = (v > 0.0f) ? v : (__expf(v) - 1.0f);      // ELU(alpha=1)
  }
}

__global__ void bias_add_inplace(float* __restrict__ x, const float* __restrict__ b,
                                 long long n, int F) {
  long long i = (long long)blockIdx.x * blockDim.x + threadIdx.x;
  long long stride = (long long)gridDim.x * blockDim.x;
  for (; i < n; i += stride) x[i] += b[i % F];
}

// ---------------------------------------------------------------------------
static inline size_t alignUp(size_t x) { return (x + 255) & ~(size_t)255; }

static void launch_gemm(const float* A, const __bf16* Bt, float* C,
                        long long M, int N, int K, hipStream_t stream) {
  const int waves = (N >= 64) ? 4 : 1;               // column tiles per block
  dim3 blk(32, waves, 1);
  dim3 grd((unsigned)(N / (16 * waves)), (unsigned)(M / 16), 1);
  if (K == 256)      gemm_wmma_bf16<256><<<grd, blk, 0, stream>>>(A, Bt, C, (int)M, N, K);
  else if (K == 128) gemm_wmma_bf16<128><<<grd, blk, 0, stream>>>(A, Bt, C, (int)M, N, K);
  else               gemm_wmma_bf16<0>  <<<grd, blk, 0, stream>>>(A, Bt, C, (int)M, N, K);
}

extern "C" void kernel_launch(void* const* d_in, const int* in_sizes, int n_in,
                              void* d_out, int out_size, void* d_ws, size_t ws_size,
                              hipStream_t stream) {
  // inputs: x, edge_index, W1, att_src1, att_dst1, b1, W2, att_src2, att_dst2, b2
  const float* x    = (const float*)d_in[0];
  const int*   ei   = (const int*)  d_in[1];         // int32 edge_index [2,E]
  const float* W1   = (const float*)d_in[2];
  const float* asp1 = (const float*)d_in[3];
  const float* adp1 = (const float*)d_in[4];
  const float* b1   = (const float*)d_in[5];
  const float* W2   = (const float*)d_in[6];
  const float* asp2 = (const float*)d_in[7];
  const float* adp2 = (const float*)d_in[8];
  const float* b2   = (const float*)d_in[9];

  const int  HC   = in_sizes[5];                     // H*C = 128
  const int  Fin  = in_sizes[2] / HC;                // 256
  const long long Nn = (long long)in_sizes[0] / Fin; // 100000
  const int  NCLS = in_sizes[9];                     // 16
  const long long E    = (long long)in_sizes[1] / 2; // 1.6M
  const long long Etot = E + Nn;                     // + self-loops
  const int  H1 = 4, C1 = HC / 4;

  // workspace carve-up
  char* ws = (char*)d_ws; size_t p = 0;
  float* hfeat1 = (float*)(ws + p); p += alignUp(sizeof(float) * Nn * HC);
  float* agg1   = (float*)(ws + p); p += alignUp(sizeof(float) * Nn * HC);
  float* ews    = (float*)(ws + p); p += alignUp(sizeof(float) * Etot * H1);
  float* as1    = (float*)(ws + p); p += alignUp(sizeof(float) * Nn * H1);
  float* ad1    = (float*)(ws + p); p += alignUp(sizeof(float) * Nn * H1);
  float* m1     = (float*)(ws + p); p += alignUp(sizeof(float) * Nn * H1);
  float* s1     = (float*)(ws + p); p += alignUp(sizeof(float) * Nn * H1);
  float* t2     = (float*)(ws + p); p += alignUp(sizeof(float) * Nn * NCLS);
  float* as2    = (float*)(ws + p); p += alignUp(sizeof(float) * Nn);
  float* ad2    = (float*)(ws + p); p += alignUp(sizeof(float) * Nn);
  float* m2     = (float*)(ws + p); p += alignUp(sizeof(float) * Nn);
  float* s2     = (float*)(ws + p); p += alignUp(sizeof(float) * Nn);
  __bf16* W1t   = (__bf16*)(ws + p); p += alignUp(sizeof(__bf16) * Fin * HC);
  __bf16* W2t   = (__bf16*)(ws + p); p += alignUp(sizeof(__bf16) * HC * NCLS);
  (void)ws_size; (void)n_in; (void)out_size;

  const int TB = 256;
  auto blocks = [&](long long total) -> int {
    long long b = (total + TB - 1) / TB;
    return (int)(b > 1048576 ? 1048576 : b);
  };
  const float NEG_INF = -__builtin_inff();

  // ---- init accumulators (every launch -> deterministic) ----
  fill_f32<<<blocks(Nn * HC), TB, 0, stream>>>(agg1, 0.0f, Nn * HC);
  fill_f32<<<blocks(Nn * H1), TB, 0, stream>>>(m1, NEG_INF, Nn * H1);
  fill_f32<<<blocks(Nn * H1), TB, 0, stream>>>(s1, 0.0f, Nn * H1);
  fill_f32<<<blocks(Nn * NCLS), TB, 0, stream>>>((float*)d_out, 0.0f, Nn * NCLS);
  fill_f32<<<blocks(Nn), TB, 0, stream>>>(m2, NEG_INF, Nn);
  fill_f32<<<blocks(Nn), TB, 0, stream>>>(s2, 0.0f, Nn);

  // ---- weights: transpose + bf16 ----
  cvt_transpose_bf16<<<(Fin * HC + TB - 1) / TB, TB, 0, stream>>>(W1, W1t, Fin, HC);
  cvt_transpose_bf16<<<(HC * NCLS + TB - 1) / TB, TB, 0, stream>>>(W2, W2t, HC, NCLS);

  // ================= Layer 1 =================
  launch_gemm(x, W1t, hfeat1, Nn, HC, Fin, stream);
  attn_coef<<<blocks(Nn * H1), TB, 0, stream>>>(hfeat1, asp1, adp1, as1, ad1, Nn, H1, C1);
  edge_logits<<<blocks(Etot * H1), TB, 0, stream>>>(ei, E, Etot, as1, ad1, ews, m1, H1);
  edge_expsum<<<blocks(Etot * H1), TB, 0, stream>>>(ei, E, Etot, m1, ews, s1, H1);
  edge_aggregate<<<blocks(Etot * HC), TB, 0, stream>>>(ei, E, Etot, hfeat1, ews, s1, agg1, H1, C1);
  bias_elu_inplace<<<blocks(Nn * HC), TB, 0, stream>>>(agg1, b1, Nn * HC, HC);

  // ================= Layer 2 =================
  launch_gemm(agg1, W2t, t2, Nn, NCLS, HC, stream);
  attn_coef<<<blocks(Nn), TB, 0, stream>>>(t2, asp2, adp2, as2, ad2, Nn, 1, NCLS);
  edge_logits<<<blocks(Etot), TB, 0, stream>>>(ei, E, Etot, as2, ad2, ews, m2, 1);
  edge_expsum<<<blocks(Etot), TB, 0, stream>>>(ei, E, Etot, m2, ews, s2, 1);
  edge_aggregate<<<blocks(Etot * NCLS), TB, 0, stream>>>(ei, E, Etot, t2, ews, s2,
                                                         (float*)d_out, 1, NCLS);
  bias_add_inplace<<<blocks(Nn * NCLS), TB, 0, stream>>>((float*)d_out, b2, Nn * NCLS, NCLS);
}
